// TriangularUpdate_10136122819092
// MI455X (gfx1250) — matrix operand order repaired and checked by reading, not compiled, not run on hardware
//
#include <hip/hip_runtime.h>
#include <hip/hip_bf16.h>

// ---------------------------------------------------------------------------
// TriangularUpdate for MI455X (gfx1250, wave32, WMMA, async LDS copies).
//   K1: LayerNorm + dual gated projection (WMMA bf16)
//   K2: 32x per-(b,head) 512^3 triangular GEMMs, double-buffered
//       GLOBAL_LOAD_ASYNC_TO_LDS staging (ASYNCcnt) + WMMA bf16
//   K3: Va @ Wo + sigmoid gate (VALU; K=16, HBM-bound output stream)
// Roofline: ~21 GFLOP vs >=536MB HBM traffic -> memory bound; bf16 operands
// halve intermediate bytes, all intermediates L2-resident (192MB).
// ---------------------------------------------------------------------------

namespace {
constexpr int kB = 2;
constexpr int kN = 512;
constexpr int kC = 128;
constexpr int kH = 8;

typedef __bf16 bf16_t;
typedef bf16_t v8bf  __attribute__((ext_vector_type(8)));
typedef bf16_t v16bf __attribute__((ext_vector_type(16)));
typedef float  v8f   __attribute__((ext_vector_type(8)));
typedef int    v4i   __attribute__((ext_vector_type(4)));

__device__ __forceinline__ float sigmoidf_(float x) {
  return 1.0f / (1.0f + __expf(-x));
}

// ---- async global->LDS (CDNA5) with sync fallback -------------------------
#if __has_builtin(__builtin_amdgcn_global_load_async_to_lds_b128)
#define HAS_ASYNC_LDS 1
#else
#define HAS_ASYNC_LDS 0
#endif

__device__ __forceinline__ void copy16_g2l(const bf16_t* g, bf16_t* l) {
#if HAS_ASYNC_LDS
  // Prototype (probe-derived): (int4 AS1* src, int4 AS3* dst, imm off, imm cpol)
  __builtin_amdgcn_global_load_async_to_lds_b128(
      (__attribute__((address_space(1))) v4i*)g,
      (__attribute__((address_space(3))) v4i*)l, 0, 0);
#else
  *(v8bf*)l = *(const v8bf*)g;  // global_load_b128 + ds_store_b128
#endif
}

__device__ __forceinline__ void wait_async_le3() {
#if HAS_ASYNC_LDS
#if __has_builtin(__builtin_amdgcn_s_wait_asynccnt)
  __builtin_amdgcn_s_wait_asynccnt(3);
#else
  asm volatile("s_wait_asynccnt 0x3" ::: "memory");
#endif
#endif
}

__device__ __forceinline__ void wait_async_0() {
#if HAS_ASYNC_LDS
#if __has_builtin(__builtin_amdgcn_s_wait_asynccnt)
  __builtin_amdgcn_s_wait_asynccnt(0);
#else
  asm volatile("s_wait_asynccnt 0x0" ::: "memory");
#endif
#endif
}

// A-fragment (16x32 bf16, row-major [m][k]). Layout per ISA 7.12.2:
// lane m=l%16; element e -> K = (e/8)*16 + (l/16)*8 + e%8.
__device__ __forceinline__ v16bf ld_frag_a(const bf16_t* base, int stride, int lane) {
  const int m = lane & 15, half = lane >> 4;
  const bf16_t* p = base + (long)m * stride + half * 8;
  v8bf lo = *(const v8bf*)(p);       // K = half*8 + 0..7
  v8bf hi = *(const v8bf*)(p + 16);  // K = 16 + half*8 + 0..7
  v16bf f;
#pragma unroll
  for (int i = 0; i < 8; ++i) { f[i] = lo[i]; f[8 + i] = hi[i]; }
  return f;
}

// B-fragment (32x16 bf16) from an N-major (B^T) buffer [n][k]:
// lane n=l%16; element e -> K = (e/8)*16 + (l/16)*8 + e%8.
__device__ __forceinline__ v16bf ld_frag_b(const bf16_t* base, int stride, int lane) {
  const int n = lane & 15, half = lane >> 4;
  const bf16_t* p = base + (long)n * stride + half * 8;
  v8bf lo = *(const v8bf*)(p);
  v8bf hi = *(const v8bf*)(p + 16);
  v16bf f;
#pragma unroll
  for (int i = 0; i < 8; ++i) { f[i] = lo[i]; f[8 + i] = hi[i]; }
  return f;
}

__device__ __forceinline__ v8f wmma_bf16(v16bf a, v16bf b, v8f c) {
  return __builtin_amdgcn_wmma_f32_16x16x32_bf16(false, a, false, b,
                                                 (short)0, c, false, false);
}

// ---------------------------------------------------------------------------
// Kernel 1: 16 rows of e per block. LN -> bf16 A-tile in LDS -> WMMA against
// Wv/We (staged transposed in LDS) -> sigmoid gating -> 4 bf16 operand
// planes laid out so K2 sees uniform A(MxK) * B^T(NxK) GEMMs:
//   Ein  [b][h][i][k] = gate(E_in)[b,i,k,h]     Vin  [b][h][j][k] = gate(V_in)[b,j,k,h]
//   EoutT[b][h][i][k] = gate(E_out)[b,k,i,h]    VoutT[b][h][j][k] = gate(V_out)[b,k,j,h]
// ---------------------------------------------------------------------------
__global__ __launch_bounds__(128) void k1_ln_proj_gate(
    const float* __restrict__ e, const float* __restrict__ mask,
    const float* __restrict__ ln_w, const float* __restrict__ ln_b,
    const float* __restrict__ Wv, const float* __restrict__ bv,
    const float* __restrict__ We, const float* __restrict__ be,
    bf16_t* __restrict__ Ein, bf16_t* __restrict__ Vin,
    bf16_t* __restrict__ EoutT, bf16_t* __restrict__ VoutT) {
  __shared__ bf16_t s_wtv[32 * 128];
  __shared__ bf16_t s_wte[32 * 128];
  __shared__ bf16_t s_eln[16 * 128];
  __shared__ float  s_lnw[kC], s_lnb[kC];
  __shared__ float  s_bv[32], s_be[32];
  __shared__ float  s_red[16][8][2];
  __shared__ float  s_v[16][32];
  __shared__ float  s_e[16][32];
  __shared__ float  s_mask[16];

  const int t = threadIdx.x;

#pragma unroll
  for (int i = 0; i < 32; ++i) {
    const int idx = t * 32 + i;  // over [128][32]
    const int k = idx >> 5, col = idx & 31;
    s_wtv[col * 128 + k] = (bf16_t)Wv[idx];
    s_wte[col * 128 + k] = (bf16_t)We[idx];
  }
  s_lnw[t] = ln_w[t];
  s_lnb[t] = ln_b[t];
  if (t < 32) { s_bv[t] = bv[t]; s_be[t] = be[t]; }

  // LayerNorm: 8 threads per row, 16 floats each.
  const long r0 = (long)blockIdx.x * 16;
  const int ri = t >> 3;
  const int sub = t & 7;
  const long r = r0 + ri;
  const float* erow = e + r * kC + sub * 16;
  float vals[16];
  float s = 0.f, ss = 0.f;
#pragma unroll
  for (int i = 0; i < 16; ++i) {
    const float x = erow[i];
    vals[i] = x; s += x; ss += x * x;
  }
  s_red[ri][sub][0] = s;
  s_red[ri][sub][1] = ss;
  if (sub == 0) s_mask[ri] = mask[r];
  __syncthreads();

  float sum = 0.f, sumsq = 0.f;
#pragma unroll
  for (int i = 0; i < 8; ++i) { sum += s_red[ri][i][0]; sumsq += s_red[ri][i][1]; }
  const float mu = sum * (1.0f / 128.0f);
  const float var = sumsq * (1.0f / 128.0f) - mu * mu;
  const float rstd = __frsqrt_rn(var + 1e-5f);
#pragma unroll
  for (int i = 0; i < 16; ++i) {
    const int c = sub * 16 + i;
    s_eln[ri * 128 + c] = (bf16_t)((vals[i] - mu) * rstd * s_lnw[c] + s_lnb[c]);
  }
  __syncthreads();

  // WMMA projections: wave w -> {V,E} x {cols 0-15, 16-31}.
  const int lane = t & 31;
  const int wave = t >> 5;
  const int matsel = wave >> 1;
  const int nb = wave & 1;
  const bf16_t* wt = matsel ? s_wte : s_wtv;
  v8f acc;
#pragma unroll
  for (int i = 0; i < 8; ++i) acc[i] = 0.f;
#pragma unroll
  for (int ks = 0; ks < 4; ++ks) {
    const v16bf a = ld_frag_a(s_eln + ks * 32, 128, lane);
    const v16bf b = ld_frag_b(wt + nb * 16 * 128 + ks * 32, 128, lane);
    acc = wmma_bf16(a, b, acc);
  }
  {
    float(*dmat)[32] = matsel ? s_e : s_v;
    const int n = lane & 15, half = lane >> 4;
#pragma unroll
    for (int rr = 0; rr < 8; ++rr) dmat[rr + half * 8][nb * 16 + n] = acc[rr];
  }
  __syncthreads();

  // Sigmoid gating + layout-aware stores.
  const int h = t & 7;
  const int ri2 = t >> 3;
  const long rr = r0 + ri2;
  const int bb = (int)(rr >> 18);
  const int p = (int)((rr >> 9) & 511);
  const int q = (int)(rr & 511);
  const float mk = s_mask[ri2];

  const float vin  = sigmoidf_(s_v[ri2][h]      + s_bv[h]      + mk) * (s_v[ri2][h + 8]  + s_bv[h + 8]);
  const float vout = sigmoidf_(s_v[ri2][h + 16] + s_bv[h + 16] + mk) * (s_v[ri2][h + 24] + s_bv[h + 24]);
  const float ein  = sigmoidf_(s_e[ri2][h]      + s_be[h]      + mk) * (s_e[ri2][h + 8]  + s_be[h + 8]);
  const float eout = sigmoidf_(s_e[ri2][h + 16] + s_be[h + 16] + mk) * (s_e[ri2][h + 24] + s_be[h + 24]);

  const long plane = (long)kN * kN;
  const long idx  = (long)(bb * kH + h) * plane + (long)p * kN + q;
  const long idxT = (long)(bb * kH + h) * plane + (long)q * kN + p;
  Ein[idx]    = (bf16_t)ein;
  Vin[idx]    = (bf16_t)vin;
  EoutT[idxT] = (bf16_t)eout;
  VoutT[idxT] = (bf16_t)vout;
}

// ---------------------------------------------------------------------------
// Kernel 2: per (b, ch) a 512x512x512 bf16 GEMM  D = A * B^T, f32 accum.
// Block = 256 threads = 8 waves; block tile 128(M) x 64(N), K staged 32 at a
// time into double-buffered LDS via async global->LDS copies (ASYNCcnt).
// LDS rows padded to 40 elements (80B) to dodge bank conflicts on the
// 16-lane fragment gathers. Output written planar [b][ch][i][j] (coalesced).
// ---------------------------------------------------------------------------
constexpr int kPad = 40;  // padded K-row stride in LDS (elements)

__global__ __launch_bounds__(256) void k2_trimul(
    const bf16_t* __restrict__ Ein, const bf16_t* __restrict__ Vin,
    const bf16_t* __restrict__ EoutT, const bf16_t* __restrict__ VoutT,
    float* __restrict__ Va) {
  __shared__ bf16_t sA[2][128 * kPad];  // 20KB
  __shared__ bf16_t sB[2][64 * kPad];   // 10KB

  const int bb = blockIdx.z;
  const int ch = blockIdx.y;          // 0..15: 0-7 "in", 8-15 "out"
  const int mtile = blockIdx.x & 3;   // 4 x 128
  const int ntile = blockIdx.x >> 2;  // 8 x 64
  const long plane = (long)kN * kN;
  const long pofs = (long)(bb * kH + (ch & 7)) * plane;
  const bf16_t* A  = (ch < 8) ? (Ein + pofs) : (EoutT + pofs);
  const bf16_t* Bm = (ch < 8) ? (Vin + pofs) : (VoutT + pofs);

  const int t = threadIdx.x, lane = t & 31, wave = t >> 5;
  const int wm = wave >> 1, wn = wave & 1;
  const int mtb = mtile * 128, ntb = ntile * 64;

  // Per-stage copies: A = 512 16B-chunks (2/thread), B = 256 (1/thread).
  const int cr = t >> 2;      // chunk row
  const int cc = (t & 3) * 8; // chunk col (elements)
  auto copy_stage = [&](int buf, int k0) {
    copy16_g2l(A + (long)(mtb + cr) * kN + k0 + cc,       &sA[buf][cr * kPad + cc]);
    copy16_g2l(A + (long)(mtb + 64 + cr) * kN + k0 + cc,  &sA[buf][(64 + cr) * kPad + cc]);
    copy16_g2l(Bm + (long)(ntb + cr) * kN + k0 + cc,      &sB[buf][cr * kPad + cc]);
  };

  v8f acc00, acc01, acc10, acc11;
#pragma unroll
  for (int i = 0; i < 8; ++i) { acc00[i] = 0.f; acc01[i] = 0.f; acc10[i] = 0.f; acc11[i] = 0.f; }

  copy_stage(0, 0);
  for (int s = 0; s < 16; ++s) {        // K = 512 in 16 stages of 32
    const int buf = s & 1;
    if (s + 1 < 16) {
      copy_stage(buf ^ 1, (s + 1) * 32);  // prefetch next stage into other buffer
      wait_async_le3();                    // stage s done; stage s+1 in flight
    } else {
      wait_async_0();
    }
    __syncthreads();

    const bf16_t* a_base = &sA[buf][wm * 32 * kPad];
    const bf16_t* b_base = &sB[buf][wn * 32 * kPad];
    const v16bf a0 = ld_frag_a(a_base, kPad, lane);
    const v16bf a1 = ld_frag_a(a_base + 16 * kPad, kPad, lane);
    const v16bf b0 = ld_frag_b(b_base, kPad, lane);
    const v16bf b1 = ld_frag_b(b_base + 16 * kPad, kPad, lane);
    acc00 = wmma_bf16(a0, b0, acc00);
    acc01 = wmma_bf16(a0, b1, acc01);
    acc10 = wmma_bf16(a1, b0, acc10);
    acc11 = wmma_bf16(a1, b1, acc11);

    __syncthreads();  // all waves done with buf before it is overwritten
  }

  // D layout: lane l, reg r -> (m = r + (l/16)*8, n = l%16). Planar output
  // Va[b][ch][i][j]: 16 lanes write 64B contiguous -> coalesced.
  const int n = lane & 15, half = lane >> 4;
  float* outp = Va + ((long)bb * 16 + ch) * plane;
  const int gi0 = mtb + wm * 32 + half * 8;
  const int gj0 = ntb + wn * 32 + n;
#pragma unroll
  for (int r = 0; r < 8; ++r) {
    outp[(long)(gi0 + r) * kN + gj0]           = acc00[r];
    outp[(long)(gi0 + r) * kN + gj0 + 16]      = acc01[r];
    outp[(long)(gi0 + 16 + r) * kN + gj0]      = acc10[r];
    outp[(long)(gi0 + 16 + r) * kN + gj0 + 16] = acc11[r];
  }
}

// ---------------------------------------------------------------------------
// Kernel 3: O = Va @ Wo + bo; out = sigmoid(O[:,:128]) * O[:,128:].
// Va is planar [b][ch][i][j]; the 16 per-row gathers are uniform-address
// broadcast loads from L2-resident planes. K=16 -> VALU, hidden under the
// 268MB coalesced output stream.
// ---------------------------------------------------------------------------
__global__ __launch_bounds__(128) void k3_outproj(
    const float* __restrict__ Va, const float* __restrict__ Wo,
    const float* __restrict__ bo, float* __restrict__ out) {
  __shared__ float s_wo[16 * 256];
  __shared__ float s_bo[256];
  const int t = threadIdx.x;
#pragma unroll
  for (int i = 0; i < 32; ++i) s_wo[i * 128 + t] = Wo[i * 128 + t];
  s_bo[t] = bo[t];
  s_bo[t + 128] = bo[t + 128];
  __syncthreads();

  const long plane = (long)kN * kN;
  const long r0 = (long)blockIdx.x * 32;
  const int bb = (int)(r0 >> 18);
  const long ij0 = r0 & (plane - 1);
  const float* vab = Va + (long)bb * 16 * plane;

#pragma unroll 2
  for (int rr = 0; rr < 32; ++rr) {
    const long ij = ij0 + rr;
    float g = s_bo[t];
    float l = s_bo[t + 128];
#pragma unroll
    for (int k = 0; k < 16; ++k) {
      const float v = vab[(long)k * plane + ij];  // uniform across block
      g = __builtin_fmaf(v, s_wo[k * 256 + t], g);
      l = __builtin_fmaf(v, s_wo[k * 256 + t + 128], l);
    }
    out[(r0 + rr) * (long)kC + t] = sigmoidf_(g) * l;
  }
}

}  // namespace

extern "C" void kernel_launch(void* const* d_in, const int* in_sizes, int n_in,
                              void* d_out, int out_size, void* d_ws, size_t ws_size,
                              hipStream_t stream) {
  (void)in_sizes; (void)n_in; (void)out_size; (void)ws_size;
  const float* e    = (const float*)d_in[0];
  const float* mask = (const float*)d_in[1];
  const float* ln_w = (const float*)d_in[2];
  const float* ln_b = (const float*)d_in[3];
  const float* Wv   = (const float*)d_in[4];
  const float* bv   = (const float*)d_in[5];
  const float* We   = (const float*)d_in[6];
  const float* be   = (const float*)d_in[7];
  const float* Wo   = (const float*)d_in[8];
  const float* bo   = (const float*)d_in[9];

  // Workspace: 4 bf16 operand planes (4 x 8.39MB) + f32 Va (33.6MB) = 64MB.
  const size_t planeE = (size_t)kB * kH * kN * kN;
  bf16_t* Ein   = (bf16_t*)d_ws;
  bf16_t* Vin   = Ein + planeE;
  bf16_t* EoutT = Vin + planeE;
  bf16_t* VoutT = EoutT + planeE;
  float*  Va    = (float*)(VoutT + planeE);

  const long rows = (long)kB * kN * kN;  // 524288
  k1_ln_proj_gate<<<(int)(rows / 16), 128, 0, stream>>>(
      e, mask, ln_w, ln_b, Wv, bv, We, be, Ein, Vin, EoutT, VoutT);

  dim3 g2(32, 16, kB);  // (M/128 * N/64, channels, batch)
  k2_trimul<<<g2, 256, 0, stream>>>(Ein, Vin, EoutT, VoutT, Va);

  k3_outproj<<<(int)(rows / 32), 128, 0, stream>>>(Va, Wo, bo, (float*)d_out);
}
